// EncoderModel_13426067767653
// MI455X (gfx1250) — matrix operand order repaired and checked by reading, not compiled, not run on hardware
//
#include <hip/hip_runtime.h>
#include <hip/hip_bf16.h>

typedef __attribute__((ext_vector_type(16))) __bf16        v16bf;
typedef __attribute__((ext_vector_type(8)))  float         v8f;
typedef __attribute__((ext_vector_type(4)))  unsigned int  v4u;
typedef __attribute__((ext_vector_type(8)))  int           v8i;
typedef __attribute__((ext_vector_type(4)))  int           v4i;

#define DEV __device__ __forceinline__

constexpr int NB      = 2;
constexpr int CIN     = 2048;
constexpr int L       = 8192;
constexpr int D       = 64;
constexpr int KW      = 5;
constexpr int NLAYERS = 10;
constexpr int NCLS    = 19;
constexpr float EPS_IN = 1e-5f;

constexpr int BT_LD = 48;   // padded bf16 LDS row, 96B = multiple of 16B
constexpr int XS_LD = 36;   // fp32 TDM tile row stride in dwords (32 + 4 pad)

#if __has_builtin(__builtin_amdgcn_tensor_load_to_lds) && \
    __has_builtin(__builtin_amdgcn_s_wait_tensorcnt)
#define USE_TDM 1
#else
#define USE_TDM 0
#endif

// ---------- WMMA fragment helpers (m/n/acc per CDNA5 ISA 7.12.2; k order contiguous 16h+e) ----------

// A-matrix 16x32 bf16 fragment from global fp32, element stride ks.
// m = lane&15 (row baked into Wrow), kh = lane>>4.
// element e -> K = kh*16 + e
DEV v16bf a_frag_g(const float* __restrict__ Wrow, int k0, int ks, int lane) {
  int kh = (lane >> 4) << 4;
  v16bf a;
#pragma unroll
  for (int e = 0; e < 8; ++e) {
    a[e]     = (__bf16)Wrow[(k0 + kh + e) * ks];
    a[e + 8] = (__bf16)Wrow[(k0 + 8 + kh + e) * ks];
  }
  return a;
}

// B-matrix 32x16 bf16 fragment from LDS stored transposed as Bt[col][k]:
// N = lane&15 (col baked into Brow), element e -> K = (lane>>4)*16 + e
DEV v16bf b_frag_lds(const __bf16* Brow, int k0, int lane) {
  int kh = (lane >> 4) << 4;
  v16bf b;
#pragma unroll
  for (int e = 0; e < 16; ++e) b[e] = Brow[k0 + kh + e];
  return b;
}

// B fragment straight out of a TDM-written fp32 tile Xs[k][XS_LD] (row-major).
DEV v16bf b_frag_f32lds(const float* Xs, int col, int lane) {
  int kh = (lane >> 4) << 4;
  v16bf b;
#pragma unroll
  for (int e = 0; e < 16; ++e) b[e] = (__bf16)Xs[(kh + e) * XS_LD + col];
  return b;
}

DEV v8f wmma_bf16(v16bf a, v16bf b, v8f c) {
  return __builtin_amdgcn_wmma_f32_16x16x32_bf16(false, a, false, b, (short)0, c,
                                                 false, false);
}

#if USE_TDM
// Issue a TDM 2-D tile load: 32 rows (stride = rs elems) x 32 fp32 cols -> LDS,
// with LDS padding of 4 dwords every 128 B (row stride becomes XS_LD dwords).
// Descriptor layout per CDNA5 ISA section 8.3/8.4; 6-arg builtin (clang-23 form):
//   (uint32x4 g0, int32x8 g1, int32x4 g2, int32x4 g3, int32x8 extra, i32 cpol)
DEV void tdm_load_tile_f32(unsigned lds_byte_off, const float* gsrc, int rs) {
  unsigned long long ga = (unsigned long long)(uintptr_t)gsrc;
  v4u g0;
  g0[0] = 1u;                                          // count=1, user mode
  g0[1] = lds_byte_off;                                // lds_addr
  g0[2] = (unsigned)(ga & 0xffffffffu);                // global_addr lo
  g0[3] = (unsigned)((ga >> 32) & 0x01ffffffu)         // global_addr[56:32]
        | (2u << 30);                                  // type = 2 ("image")
  unsigned d0 = (2u << 16)                             // data_size = 4 bytes
              | (1u << 20)                             // pad_enable
              | (4u << 22)                             // pad_interval: 16 x 8B = 128B
              | (3u << 25);                            // pad_amount: 4 dwords
  unsigned td0 = (unsigned)rs;                         // tensor_dim0
  unsigned td1 = (unsigned)CIN;                        // tensor_dim1
  v8i g1;
  g1[0] = (int)d0;
  g1[1] = (int)((td0 & 0xffffu) << 16);                // [63:48] = td0 lo16
  g1[2] = (int)((td0 >> 16) | ((td1 & 0xffffu) << 16));
  g1[3] = (int)((td1 >> 16) | (32u << 16));            // tile_dim0 = 32
  g1[4] = (int)32u;                                    // tile_dim1 = 32, tile_dim2 = 0
  g1[5] = (int)(unsigned)rs;                           // tensor_dim0_stride lo32
  g1[6] = 0;                                           // stride hi16, dim1_stride lo
  g1[7] = 0;
  v4i z4 = {0, 0, 0, 0};
  v8i z8 = {0, 0, 0, 0, 0, 0, 0, 0};
  __builtin_amdgcn_tensor_load_to_lds(g0, g1, z4, z4, z8, 0);
}
#endif

// ---------------------- conv_in: h = W(64x2048) * x_n ----------------------
// grid (L/32, NB), block 256 (8 waves). Wave (wo,wl) owns a 16(o)x16(l) tile.
// TDM double-buffers the 32x32 fp32 x-tile Global->LDS while waves run WMMA.
__global__ __launch_bounds__(256) void k_conv_in(
    const float* __restrict__ x, const float* __restrict__ w,
    const float* __restrict__ b, float* __restrict__ h) {
  const int l0   = blockIdx.x * 32;
  const int n    = blockIdx.y;
  const int tid  = threadIdx.x;
  const int lane = tid & 31, wave = tid >> 5;
  const int wo = wave & 3, wl = wave >> 2;

  const float* xn = x + (size_t)n * CIN * L;
  const float* Wrow = w + (size_t)(wo * 16 + (lane & 15)) * CIN;
  v8f acc = {};

#if USE_TDM
  __shared__ __align__(16) float Xs[2][32 * XS_LD];
  const unsigned xoff0 = (unsigned)(uintptr_t)&Xs[0][0];
  const unsigned xoff1 = (unsigned)(uintptr_t)&Xs[1][0];
  constexpr int NSTEP = CIN / 32;

  if (tid < 32) tdm_load_tile_f32(xoff0, xn + l0, L);          // prime buffer 0

  for (int it = 0; it < NSTEP; ++it) {
    const int k0 = it * 32;
    __builtin_amdgcn_s_wait_tensorcnt(0);    // issuing wave drains; others NOP
    __syncthreads();                         // publish buffer it&1
    if (tid < 32 && it + 1 < NSTEP)          // async prefetch of next tile
      tdm_load_tile_f32((it & 1) ? xoff0 : xoff1,
                        xn + (size_t)(k0 + 32) * L + l0, L);
    if (k0 + 32 < CIN)
      __builtin_prefetch(&Wrow[k0 + 32 + lane], 0, 1);         // L2 warm weights

    v16bf a  = a_frag_g(Wrow, k0, 1, lane);
    v16bf bb = b_frag_f32lds(&Xs[it & 1][0], wl * 16 + (lane & 15), lane);
    acc = wmma_bf16(a, bb, acc);
    __syncthreads();                         // reads done before buffer reuse
  }
#else
  __shared__ __align__(16) __bf16 Bt[32][BT_LD];
  for (int k0 = 0; k0 < CIN; k0 += 32) {
#pragma unroll
    for (int i = 0; i < 4; ++i) {            // stage 32(k) x 32(l), transposed
      int idx = tid + 256 * i;
      int kk = idx >> 5, ll = idx & 31;
      Bt[ll][kk] = (__bf16)xn[(size_t)(k0 + kk) * L + l0 + ll];
    }
    if (k0 + 32 < CIN)
      __builtin_prefetch(&xn[(size_t)(k0 + 32 + wave * 4) * L + l0 + lane], 0, 1);
    __syncthreads();
    v16bf a  = a_frag_g(Wrow, k0, 1, lane);
    v16bf bb = b_frag_lds(&Bt[wl * 16 + (lane & 15)][0], 0, lane);
    acc = wmma_bf16(a, bb, acc);
    __syncthreads();
  }
#endif

  const int nn = lane & 15, kh = lane >> 4;
  const int lc = l0 + wl * 16 + nn;
#pragma unroll
  for (int r = 0; r < 8; ++r) {
    int o = wo * 16 + r + 8 * kh;            // D-frag: M = r + 8*kh
    h[((size_t)n * D + o) * L + lc] = acc[r] + b[o];
  }
}

// ---------- per-layer: q,k,v = Wqkv*h ; out1 = dilated conv (5 taps) ----------
__global__ __launch_bounds__(256) void k_qkv_conv(
    const float* __restrict__ h,
    const float* __restrict__ wq, const float* __restrict__ bq,
    const float* __restrict__ wk, const float* __restrict__ bk,
    const float* __restrict__ wv, const float* __restrict__ bv,
    const float* __restrict__ cw, const float* __restrict__ cb, int dil,
    float* __restrict__ q, float* __restrict__ k, float* __restrict__ v,
    float* __restrict__ o1) {
  __shared__ __align__(16) __bf16 Bt[KW][32][BT_LD];   // per-tap shifted h tiles
  const int l0   = blockIdx.x * 32;
  const int n    = blockIdx.y;
  const int tid  = threadIdx.x;
  const int lane = tid & 31, wave = tid >> 5;
  const int wo = wave & 3, wl = wave >> 2;
  const float* hn = h + (size_t)n * D * L;

  v8f aq = {}, ak = {}, av = {}, ac = {};

  for (int c0 = 0; c0 < D; c0 += 32) {
#pragma unroll
    for (int t = 0; t < KW; ++t) {
      int off = (t - KW / 2) * dil;
#pragma unroll
      for (int i = 0; i < 4; ++i) {
        int idx = tid + 256 * i;
        int kk = idx >> 5, ll = idx & 31;
        int col = l0 + ll + off;
        float val = ((unsigned)col < (unsigned)L)
                        ? hn[(size_t)(c0 + kk) * L + col] : 0.f;
        Bt[t][ll][kk] = (__bf16)val;
      }
    }
    __syncthreads();

    const int arow = wo * 16 + (lane & 15);
    v16bf bc = b_frag_lds(&Bt[KW / 2][wl * 16 + (lane & 15)][0], 0, lane);
    aq = wmma_bf16(a_frag_g(wq + (size_t)arow * D, c0, 1, lane), bc, aq);
    ak = wmma_bf16(a_frag_g(wk + (size_t)arow * D, c0, 1, lane), bc, ak);
    av = wmma_bf16(a_frag_g(wv + (size_t)arow * D, c0, 1, lane), bc, av);
#pragma unroll
    for (int t = 0; t < KW; ++t) {           // tap t: W_t[o][c] = cw[(o*D+c)*KW + t]
      v16bf at = a_frag_g(cw + (size_t)arow * D * KW + t, c0, KW, lane);
      v16bf bt = b_frag_lds(&Bt[t][wl * 16 + (lane & 15)][0], 0, lane);
      ac = wmma_bf16(at, bt, ac);
    }
    __syncthreads();
  }

  const int nn = lane & 15, kh = lane >> 4;
  const int lc = l0 + wl * 16 + nn;
#pragma unroll
  for (int r = 0; r < 8; ++r) {
    int o = wo * 16 + r + 8 * kh;
    size_t idx = ((size_t)n * D + o) * L + lc;
    q[idx]  = aq[r] + bq[o];
    k[idx]  = ak[r] + bk[o];
    v[idx]  = av[r] + bv[o];
    o1[idx] = ac[r] + cb[o];
  }
}

// -------- windowed attention + s = out1+out2 + instance-norm partials --------
__global__ __launch_bounds__(256) void k_attn(
    const float* __restrict__ q, const float* __restrict__ kd,
    const float* __restrict__ v, const float* __restrict__ o1,
    float* __restrict__ s, float* __restrict__ sum, float* __restrict__ sumsq,
    int dil) {
  const int n = blockIdx.y;
  const int l = blockIdx.x * 256 + threadIdx.x;
  const int lane = threadIdx.x & 31;
  const size_t base = (size_t)n * D * L + l;

  int off[KW]; bool ok[KW]; float dot[KW];
#pragma unroll
  for (int t = 0; t < KW; ++t) {
    int col = l + (t - KW / 2) * dil;
    ok[t]  = (unsigned)col < (unsigned)L;
    off[t] = ok[t] ? (t - KW / 2) * dil : 0;
    dot[t] = 0.f;
  }
#pragma unroll 4
  for (int c = 0; c < D; ++c) {
    float qv = q[base + (size_t)c * L];
#pragma unroll
    for (int t = 0; t < KW; ++t) dot[t] += qv * kd[base + (size_t)c * L + off[t]];
  }
  float mx = -1e30f;
#pragma unroll
  for (int t = 0; t < KW; ++t) { dot[t] *= 0.125f; if (ok[t] && dot[t] > mx) mx = dot[t]; }
  float den = 0.f, att[KW];
#pragma unroll
  for (int t = 0; t < KW; ++t) { att[t] = ok[t] ? __expf(dot[t] - mx) : 0.f; den += att[t]; }
  float inv = 1.f / den;
#pragma unroll
  for (int t = 0; t < KW; ++t) att[t] *= inv;

  for (int c = 0; c < D; ++c) {
    float acc = 0.f;
#pragma unroll
    for (int t = 0; t < KW; ++t) acc += att[t] * v[base + (size_t)c * L + off[t]];
    float sv = o1[base + (size_t)c * L] + acc;
    s[base + (size_t)c * L] = sv;
    float r1 = sv, r2 = sv * sv;             // wave32 reduction for IN stats
#pragma unroll
    for (int m = 16; m; m >>= 1) { r1 += __shfl_xor(r1, m, 32); r2 += __shfl_xor(r2, m, 32); }
    if (lane == 0) {
      atomicAdd(&sum[n * D + c], r1);
      atomicAdd(&sumsq[n * D + c], r2);
    }
  }
}

// ---------- instance-norm + FFN (two 64x64 WMMA GEMMs) + residual ----------
__global__ __launch_bounds__(256) void k_norm_ffn(
    float* __restrict__ h, const float* __restrict__ s,
    const float* __restrict__ sum, const float* __restrict__ sumsq,
    const float* __restrict__ w1, const float* __restrict__ b1,
    const float* __restrict__ w2, const float* __restrict__ b2) {
  __shared__ __align__(16) __bf16 Bt[32][D + 16];   // normalized h, [l][c]
  __shared__ __align__(16) __bf16 Tt[32][D + 16];   // relu(FFN1),   [l][o]
  const int l0   = blockIdx.x * 32;
  const int n    = blockIdx.y;
  const int tid  = threadIdx.x;
  const int lane = tid & 31, wave = tid >> 5;
  const int wo = wave & 3, wl = wave >> 2;
  const float invL = 1.f / (float)L;

#pragma unroll
  for (int i = 0; i < 8; ++i) {
    int idx = tid + 256 * i;
    int c = idx >> 5, ll = idx & 31;
    float mean = sum[n * D + c] * invL;
    float var  = sumsq[n * D + c] * invL - mean * mean;
    float rs   = rsqrtf(var + EPS_IN);
    float sv   = s[((size_t)n * D + c) * L + l0 + ll];
    Bt[ll][c] = (__bf16)((sv - mean) * rs);
  }
  __syncthreads();

  v8f acc1 = {};
#pragma unroll
  for (int c0 = 0; c0 < D; c0 += 32) {
    v16bf a = a_frag_g(w1 + (size_t)(wo * 16 + (lane & 15)) * D, c0, 1, lane);
    v16bf b = b_frag_lds(&Bt[wl * 16 + (lane & 15)][0], c0, lane);
    acc1 = wmma_bf16(a, b, acc1);
  }
  const int nn = lane & 15, kh = lane >> 4;
#pragma unroll
  for (int r = 0; r < 8; ++r) {
    int o = wo * 16 + r + 8 * kh;
    Tt[wl * 16 + nn][o] = (__bf16)fmaxf(acc1[r] + b1[o], 0.f);
  }
  __syncthreads();

  v8f acc2 = {};
#pragma unroll
  for (int c0 = 0; c0 < D; c0 += 32) {
    v16bf a = a_frag_g(w2 + (size_t)(wo * 16 + (lane & 15)) * D, c0, 1, lane);
    v16bf b = b_frag_lds(&Tt[wl * 16 + (lane & 15)][0], c0, lane);
    acc2 = wmma_bf16(a, b, acc2);
  }
  const int lc = l0 + wl * 16 + nn;
#pragma unroll
  for (int r = 0; r < 8; ++r) {
    int o = wo * 16 + r + 8 * kh;
    size_t idx = ((size_t)n * D + o) * L + lc;
    h[idx] = h[idx] + acc2[r] + b2[o];       // residual, in-place
  }
}

// ------------------------------- conv_out 64->19 -------------------------------
__global__ __launch_bounds__(256) void k_conv_out(
    const float* __restrict__ h, const float* __restrict__ w,
    const float* __restrict__ b, float* __restrict__ out) {
  const int gid = blockIdx.x * 256 + threadIdx.x;
  const int n = gid / L, l = gid % L;
  float acc[NCLS];
#pragma unroll
  for (int o = 0; o < NCLS; ++o) acc[o] = b[o];
  for (int c = 0; c < D; ++c) {
    float hv = h[((size_t)n * D + c) * L + l];
#pragma unroll
    for (int o = 0; o < NCLS; ++o) acc[o] += w[o * D + c] * hv;
  }
#pragma unroll
  for (int o = 0; o < NCLS; ++o)
    out[((size_t)n * NCLS + o) * L + l] = acc[o];
}

// ----------------------------------- host -----------------------------------
extern "C" void kernel_launch(void* const* d_in, const int* in_sizes, int n_in,
                              void* d_out, int out_size, void* d_ws, size_t ws_size,
                              hipStream_t stream) {
  (void)in_sizes; (void)n_in; (void)out_size; (void)ws_size;
  const float* x   = (const float*)d_in[0];
  const float* ciw = (const float*)d_in[1];
  const float* cib = (const float*)d_in[2];
  const float* cow = (const float*)d_in[3];
  const float* cob = (const float*)d_in[4];

  char* ws = (char*)d_ws;
  const size_t HB = (size_t)NB * D * L * sizeof(float);   // 4 MiB
  float* h   = (float*)(ws + 0 * HB);
  float* q   = (float*)(ws + 1 * HB);
  float* k   = (float*)(ws + 2 * HB);
  float* v   = (float*)(ws + 3 * HB);
  float* o1  = (float*)(ws + 4 * HB);
  float* s   = (float*)(ws + 5 * HB);
  float* sum = (float*)(ws + 6 * HB);     // NB*D sums then NB*D sumsqs
  float* ssq = sum + NB * D;

  const dim3 blk(256);
  const dim3 grd(L / 32, NB);

  k_conv_in<<<grd, blk, 0, stream>>>(x, ciw, cib, h);

  for (int i = 0; i < NLAYERS; ++i) {
    const float* cw = (const float*)d_in[5 + i * 12 + 0];
    const float* cb = (const float*)d_in[5 + i * 12 + 1];
    const float* wq = (const float*)d_in[5 + i * 12 + 2];
    const float* bq = (const float*)d_in[5 + i * 12 + 3];
    const float* wk = (const float*)d_in[5 + i * 12 + 4];
    const float* bk = (const float*)d_in[5 + i * 12 + 5];
    const float* wv = (const float*)d_in[5 + i * 12 + 6];
    const float* bv = (const float*)d_in[5 + i * 12 + 7];
    const float* w1 = (const float*)d_in[5 + i * 12 + 8];
    const float* b1 = (const float*)d_in[5 + i * 12 + 9];
    const float* w2 = (const float*)d_in[5 + i * 12 + 10];
    const float* b2 = (const float*)d_in[5 + i * 12 + 11];
    const int dil = 1 << i;

    (void)hipMemsetAsync(sum, 0, (size_t)2 * NB * D * sizeof(float), stream);
    k_qkv_conv<<<grd, blk, 0, stream>>>(h, wq, bq, wk, bk, wv, bv, cw, cb, dil,
                                        q, k, v, o1);
    k_attn<<<dim3(L / 256, NB), blk, 0, stream>>>(q, k, v, o1, s, sum, ssq, dil);
    k_norm_ffn<<<grd, blk, 0, stream>>>(h, s, sum, ssq, w1, b1, w2, b2);
  }

  k_conv_out<<<dim3(NB * L / 256), blk, 0, stream>>>(h, cow, cob, (float*)d_out);
}
